// AdrSelection_61778809585742
// MI455X (gfx1250) — matrix operand order repaired and checked by reading, not compiled, not run on hardware
//
#include <hip/hip_runtime.h>
#include <hip/hip_bf16.h>

// ---------------------------------------------------------------------------
// Sizes (fixed by the reference)
// ---------------------------------------------------------------------------
#define Bq   64
#define Wq   10
#define Sq   32
#define Eq   128
#define Hq   256
#define Uq   256
#define Rq   10
// gates = 3*U = 768, enc input 2H = 512, spk/adr input 2H+U = 768

typedef __attribute__((ext_vector_type(16))) _Float16 v16h;
typedef __attribute__((ext_vector_type(8)))  _Float16 v8h;
typedef __attribute__((ext_vector_type(8)))  float    v8f;

// ---------------------------------------------------------------------------
// WMMA fragment helpers (layouts per CDNA5 ISA §7.12.2, wave32)
// A: 16x32 f16, lane L<16 holds row M=L, K={kb..kb+7, 16+kb..16+kb+7}, kb=(L>>4)*8
// B: 32x16 f16, lane holds col N=(L&15), K={kb..kb+7, 16+kb..16+kb+7}, kb=(L>>4)*8
// C/D 16x16 f32: elem e -> row = e + (L>=16 ? 8:0), col = L&15
// ---------------------------------------------------------------------------
__device__ __forceinline__ v16h load_a_frag(const _Float16* __restrict__ A,
                                            int lda, int row0, int k0, int lane) {
  const _Float16* p = A + (long)(row0 + (lane & 15)) * lda + k0 + ((lane >> 4) << 3);
  v8h lo = *(const v8h*)p;
  v8h hi = *(const v8h*)(p + 16);
  v16h a;
#pragma unroll
  for (int i = 0; i < 8; ++i) { a[i] = lo[i]; a[i + 8] = hi[i]; }
  return a;
}

__device__ __forceinline__ v16h load_b_frag(const _Float16* __restrict__ Wt,
                                            int ldw, int n0, int k0, int lane) {
  // B[k][n] = Wt[n][k]  (Wt is (N x K) row-major: out = in @ Wt^T)
  const _Float16* p = Wt + (long)(n0 + (lane & 15)) * ldw + k0 + ((lane >> 4) << 3);
  v8h lo = *(const v8h*)p;
  v8h hi = *(const v8h*)(p + 16);
  v16h b;
#pragma unroll
  for (int i = 0; i < 8; ++i) { b[i] = lo[i]; b[i + 8] = hi[i]; }
  return b;
}

__device__ __forceinline__ float sigmoidf_(float x) { return 1.f / (1.f + expf(-x)); }

// ---------------------------------------------------------------------------
// Generic GEMM: C(MxN, f16) = A(MxK, f16) @ W(NxK, f16)^T + bias ; optional tanh
// grid.x = M/16, block = 256 (8 waves); wave handles N-tiles wid, wid+8, ...
// ---------------------------------------------------------------------------
__global__ void gemm_bias_f16(const _Float16* __restrict__ A,
                              const _Float16* __restrict__ Wt,
                              const float* __restrict__ bias,
                              _Float16* __restrict__ C,
                              int M, int N, int K, int act) {
  const int row0 = blockIdx.x << 4;
  const int wid  = threadIdx.x >> 5;
  const int lane = threadIdx.x & 31;
  const int ntiles = N >> 4;
  for (int nt = wid; nt < ntiles; nt += 8) {
    const int n0 = nt << 4;
    v8f acc;
    const float bv = bias ? bias[n0 + (lane & 15)] : 0.f;
#pragma unroll
    for (int e = 0; e < 8; ++e) acc[e] = bv;
    for (int k0 = 0; k0 < K; k0 += 32) {
      v16h a = load_a_frag(A, K, row0, k0, lane);
      v16h b = load_b_frag(Wt, K, n0, k0, lane);
      acc = __builtin_amdgcn_wmma_f32_16x16x32_f16(false, a, false, b,
                                                   (short)0, acc, false, false);
    }
    const int col = lane & 15;
    const int rb  = (lane >> 4) << 3;
#pragma unroll
    for (int e = 0; e < 8; ++e) {
      float v = acc[e];
      if (act) v = tanhf(v);
      C[(long)(row0 + rb + e) * N + n0 + col] = (_Float16)v;
    }
  }
}

// ---------------------------------------------------------------------------
// Fused GRU step: gh = H @ Whh^T + bhh via WMMA, then gate math with
// precomputed gi (Gi = x@Wih^T + bih), optional length mask, in-place H update.
// U = 256 fixed. grid.x = M/16, block = 256. Wave w owns u-tile groups
// {w, w+8}; for each it computes the MATCHING r/z/n tiles so the gate combine
// needs no cross-lane traffic. Barrier separates reads of old H from writes.
// ---------------------------------------------------------------------------
__global__ void gru_step(const _Float16* __restrict__ Gi, long gi_stride,
                         _Float16* __restrict__ H16, float* __restrict__ H32,
                         const _Float16* __restrict__ Whh,  // (768 x 256) f16
                         const float* __restrict__ bhh,     // 768
                         const int* __restrict__ lengths, int tval,
                         int M) {
  const int row0 = blockIdx.x << 4;
  const int wid  = threadIdx.x >> 5;
  const int lane = threadIdx.x & 31;
  const int col  = lane & 15;
  const int rb   = (lane >> 4) << 3;

  v8f accR[2], accZ[2], accN[2];
#pragma unroll
  for (int i = 0; i < 2; ++i) {
    const int n0 = (wid + i * 8) << 4;
    const float br = bhh[n0 + col];
    const float bz = bhh[256 + n0 + col];
    const float bn = bhh[512 + n0 + col];
#pragma unroll
    for (int e = 0; e < 8; ++e) { accR[i][e] = br; accZ[i][e] = bz; accN[i][e] = bn; }
  }
  for (int k0 = 0; k0 < 256; k0 += 32) {
    v16h a = load_a_frag(H16, 256, row0, k0, lane);
#pragma unroll
    for (int i = 0; i < 2; ++i) {
      const int n0 = (wid + i * 8) << 4;
      accR[i] = __builtin_amdgcn_wmma_f32_16x16x32_f16(false, a, false,
                  load_b_frag(Whh, 256, n0, k0, lane), (short)0, accR[i], false, false);
      accZ[i] = __builtin_amdgcn_wmma_f32_16x16x32_f16(false, a, false,
                  load_b_frag(Whh, 256, 256 + n0, k0, lane), (short)0, accZ[i], false, false);
      accN[i] = __builtin_amdgcn_wmma_f32_16x16x32_f16(false, a, false,
                  load_b_frag(Whh, 256, 512 + n0, k0, lane), (short)0, accN[i], false, false);
    }
  }
  __syncthreads();   // all waves finished reading old H16 for this row block
#pragma unroll
  for (int i = 0; i < 2; ++i) {
    const int u = ((wid + i * 8) << 4) + col;
#pragma unroll
    for (int e = 0; e < 8; ++e) {
      const int m = row0 + rb + e;
      const _Float16* gir = Gi + (long)m * gi_stride;
      const float gi_r = (float)gir[u];
      const float gi_z = (float)gir[256 + u];
      const float gi_n = (float)gir[512 + u];
      const float hold = H32[(long)m * 256 + u];
      const float r = sigmoidf_(gi_r + accR[i][e]);
      const float z = sigmoidf_(gi_z + accZ[i][e]);
      const float n = tanhf(gi_n + r * accN[i][e]);
      float hnew = (1.f - z) * n + z * hold;
      if (lengths && !(tval < lengths[m])) hnew = hold;
      H32[(long)m * 256 + u] = hnew;
      H16[(long)m * 256 + u] = (_Float16)hnew;
    }
  }
}

// ---------------------------------------------------------------------------
// Small helper kernels
// ---------------------------------------------------------------------------
__global__ void cvt_f32_f16(const float* __restrict__ s, _Float16* __restrict__ d, long n) {
  long i = (long)blockIdx.x * blockDim.x + threadIdx.x;
  if (i < n) d[i] = (_Float16)s[i];
}

__global__ void embed_gather(const int* __restrict__ dig, const float* __restrict__ emb,
                             _Float16* __restrict__ x16) {
  long i = (long)blockIdx.x * blockDim.x + threadIdx.x;
  const long total = (long)Bq * Wq * Sq * Eq;
  if (i >= total) return;
  long ns = i >> 7;           // (n,s) pair
  int  e  = (int)(i & 127);
  int  tok = dig[ns];
  x16[i] = (_Float16)emb[(long)tok * Eq + e];
}

__device__ __forceinline__ int nth_other(int spk, int adr, int o) {
  int cnt = -1;
#pragma unroll
  for (int id = 0; id < Rq; ++id)
    if (id != spk && id != adr) { if (++cnt == o) return id; }
  return 0;
}

// Per-interaction-step gather: builds X_spk/X_adr/X_oth/X_hd (f16) and gathers
// H_spk/H_adr/H_oth (f16 + f32 master) from A.
__global__ void inter_gather(const float* __restrict__ hF, const float* __restrict__ hB,
                             const float* __restrict__ A32, const int* __restrict__ users,
                             int w,
                             _Float16* __restrict__ Xspk, _Float16* __restrict__ Xadr,
                             _Float16* __restrict__ Xoth, _Float16* __restrict__ Xhd,
                             _Float16* __restrict__ Hspk16, float* __restrict__ Hspk32,
                             _Float16* __restrict__ Hadr16, float* __restrict__ Hadr32,
                             _Float16* __restrict__ Hoth16, float* __restrict__ Hoth32) {
  const long T_XSA = 64 * 768, T_XOTH = 512 * 512, T_XHD = 64 * 512;
  const long T_H = 64 * 256, T_HOTH = 512 * 256;
  long idx = (long)blockIdx.x * blockDim.x + threadIdx.x;
  const long total = 2 * T_XSA + T_XOTH + T_XHD + 2 * T_H + T_HOTH;
  if (idx >= total) return;

#define EH(b, c) (((c) < 256) ? hF[((b) * Wq + w) * 256 + (c)] \
                              : hB[((b) * Wq + w) * 256 + ((c) - 256)])
  if (idx < T_XSA) {                       // X_spk = [eh || A[adr]]
    int b = (int)(idx / 768), c = (int)(idx % 768);
    int adr = users[(b * Wq + w) * 2 + 1];
    float v = (c < 512) ? EH(b, c) : A32[(b * Rq + adr) * 256 + (c - 512)];
    Xspk[idx] = (_Float16)v; return;
  }
  idx -= T_XSA;
  if (idx < T_XSA) {                       // X_adr = [eh || A[spk]]
    int b = (int)(idx / 768), c = (int)(idx % 768);
    int spk = users[(b * Wq + w) * 2 + 0];
    float v = (c < 512) ? EH(b, c) : A32[(b * Rq + spk) * 256 + (c - 512)];
    Xadr[idx] = (_Float16)v; return;
  }
  idx -= T_XSA;
  if (idx < T_XOTH) {                      // X_oth: eh broadcast over 8 others
    int row = (int)(idx / 512), c = (int)(idx % 512);
    int b = row >> 3;
    Xoth[idx] = (_Float16)EH(b, c); return;
  }
  idx -= T_XOTH;
  if (idx < T_XHD) {                       // X_hd = eh
    int b = (int)(idx / 512), c = (int)(idx % 512);
    Xhd[idx] = (_Float16)EH(b, c); return;
  }
  idx -= T_XHD;
  if (idx < T_H) {                         // H_spk
    int b = (int)(idx / 256), u = (int)(idx % 256);
    int spk = users[(b * Wq + w) * 2 + 0];
    float v = A32[(b * Rq + spk) * 256 + u];
    Hspk32[idx] = v; Hspk16[idx] = (_Float16)v; return;
  }
  idx -= T_H;
  if (idx < T_H) {                         // H_adr
    int b = (int)(idx / 256), u = (int)(idx % 256);
    int adr = users[(b * Wq + w) * 2 + 1];
    float v = A32[(b * Rq + adr) * 256 + u];
    Hadr32[idx] = v; Hadr16[idx] = (_Float16)v; return;
  }
  idx -= T_H;
  {                                        // H_oth
    int row = (int)(idx / 256), u = (int)(idx % 256);
    int b = row >> 3, o = row & 7;
    int spk = users[(b * Wq + w) * 2 + 0];
    int adr = users[(b * Wq + w) * 2 + 1];
    int oid = nth_other(spk, adr, o);
    float v = A32[(b * Rq + oid) * 256 + u];
    Hoth32[idx] = v; Hoth16[idx] = (_Float16)v;
  }
#undef EH
}

__global__ void inter_scatter(float* __restrict__ A32, const int* __restrict__ users, int w,
                              const float* __restrict__ Hspk32,
                              const float* __restrict__ Hadr32,
                              const float* __restrict__ Hoth32) {
  const long T_H = 64 * 256;
  long idx = (long)blockIdx.x * blockDim.x + threadIdx.x;
  if (idx >= 2 * T_H + 512 * 256) return;
  if (idx < T_H) {
    int b = (int)(idx / 256), u = (int)(idx % 256);
    int spk = users[(b * Wq + w) * 2 + 0];
    A32[(b * Rq + spk) * 256 + u] = Hspk32[idx]; return;
  }
  idx -= T_H;
  if (idx < T_H) {
    int b = (int)(idx / 256), u = (int)(idx % 256);
    int adr = users[(b * Wq + w) * 2 + 1];
    A32[(b * Rq + adr) * 256 + u] = Hadr32[idx]; return;
  }
  idx -= T_H;
  {
    int row = (int)(idx / 256), u = (int)(idx % 256);
    int b = row >> 3, o = row & 7;
    int spk = users[(b * Wq + w) * 2 + 0];
    int adr = users[(b * Wq + w) * 2 + 1];
    int oid = nth_other(spk, adr, o);
    A32[(b * Rq + oid) * 256 + u] = Hoth32[idx];
  }
}

__global__ void fc_gather(const float* __restrict__ A32, const float* __restrict__ hHd32,
                          const int* __restrict__ resp, _Float16* __restrict__ Xfc) {
  long i = (long)blockIdx.x * blockDim.x + threadIdx.x;
  if (i >= 64 * 512) return;
  int b = (int)(i >> 9), c = (int)(i & 511);
  float v = (c < 256) ? A32[(b * Rq + resp[b]) * 256 + c] : hHd32[b * 256 + (c - 256)];
  Xfc[i] = (_Float16)v;
}

__global__ void final_scores(const float* __restrict__ A32,
                             const _Float16* __restrict__ conc,
                             const int* __restrict__ resp,
                             const float* __restrict__ fc2W,
                             const float* __restrict__ fc2b,
                             float* __restrict__ out) {
  int b = blockIdx.x;
  __shared__ float sc[Rq - 1];
  int t = threadIdx.x;
  int rsp = resp[b];
  if (t < Rq - 1) {
    int pid = (t < rsp) ? t : t + 1;            // ids != responder, ascending
    const float* av = A32 + (long)(b * Rq + pid) * 256;
    const _Float16* cv = conc + (long)b * 256;
    float dot = 0.f;
    for (int u = 0; u < 256; ++u) dot += av[u] * (float)cv[u];
    sc[t] = dot;
  }
  __syncthreads();
  if (t < Rq - 1) {
    float o = fc2b[t];
#pragma unroll
    for (int r = 0; r < Rq - 1; ++r) o += sc[r] * fc2W[t * (Rq - 1) + r];
    out[b * (Rq - 1) + t] = o;
  }
}

__global__ void copy_f32(const float* __restrict__ s, float* __restrict__ d, long n) {
  long i = (long)blockIdx.x * blockDim.x + threadIdx.x;
  if (i < n) d[i] = s[i];
}

// ---------------------------------------------------------------------------
// Host driver
// ---------------------------------------------------------------------------
static inline long cdivl(long a, long b) { return (a + b - 1) / b; }

extern "C" void kernel_launch(void* const* d_in, const int* in_sizes, int n_in,
                              void* d_out, int out_size, void* d_ws, size_t ws_size,
                              hipStream_t stream) {
  (void)in_sizes; (void)n_in; (void)out_size; (void)ws_size;
  const int*   dig   = (const int*)d_in[0];
  const int*   lens  = (const int*)d_in[1];
  const int*   users = (const int*)d_in[2];
  const int*   resp  = (const int*)d_in[3];
  const float* emb   = (const float*)d_in[4];
  const float* f32w[29];
  for (int i = 0; i < 29; ++i) f32w[i] = (const float*)d_in[5 + i];
  // index map into f32w: 0 fWih 1 fWhh 2 fbih 3 fbhh 4 bWih 5 bWhh 6 bbih 7 bbhh
  // 8 sWih 9 sWhh 10 sbih 11 sbhh 12 aWih 13 aWhh 14 abih 15 abhh
  // 16 oWih 17 oWhh 18 obih 19 obhh 20 hWih 21 hWhh 22 hbih 23 hbhh
  // 24 fc1W 25 fc1b 26 fc2W 27 fc2b
  float* out = (float*)d_out;

  char* p = (char*)d_ws;
  auto alloc = [&](size_t bytes) -> void* {
    void* r = (void*)p; p += (bytes + 255) & ~(size_t)255; return r;
  };
  const long NBW = (long)Bq * Wq;           // 640
  const long NTS = NBW * Sq;                // 20480

  _Float16* x16  = (_Float16*)alloc(NTS * Eq * 2);
  _Float16* GiF  = (_Float16*)alloc(NTS * 768 * 2);
  _Float16* GiB  = (_Float16*)alloc(NTS * 768 * 2);
  _Float16* wfih = (_Float16*)alloc(768 * 128 * 2);
  _Float16* wfhh = (_Float16*)alloc(768 * 256 * 2);
  _Float16* wbih = (_Float16*)alloc(768 * 128 * 2);
  _Float16* wbhh = (_Float16*)alloc(768 * 256 * 2);
  _Float16* wsih = (_Float16*)alloc(768 * 768 * 2);
  _Float16* wshh = (_Float16*)alloc(768 * 256 * 2);
  _Float16* waih = (_Float16*)alloc(768 * 768 * 2);
  _Float16* wahh = (_Float16*)alloc(768 * 256 * 2);
  _Float16* woih = (_Float16*)alloc(768 * 512 * 2);
  _Float16* wohh = (_Float16*)alloc(768 * 256 * 2);
  _Float16* whih = (_Float16*)alloc(768 * 512 * 2);
  _Float16* whhh = (_Float16*)alloc(768 * 256 * 2);
  _Float16* wfc1 = (_Float16*)alloc(256 * 512 * 2);
  _Float16* hF16 = (_Float16*)alloc(NBW * 256 * 2);
  _Float16* hB16 = (_Float16*)alloc(NBW * 256 * 2);
  float*    hF32 = (float*)alloc(NBW * 256 * 4);
  float*    hB32 = (float*)alloc(NBW * 256 * 4);
  float*    A32  = (float*)alloc((long)Bq * Rq * 256 * 4);
  _Float16* Xspk = (_Float16*)alloc(64 * 768 * 2);
  _Float16* Xadr = (_Float16*)alloc(64 * 768 * 2);
  _Float16* Xoth = (_Float16*)alloc(512 * 512 * 2);
  _Float16* Xhd  = (_Float16*)alloc(64 * 512 * 2);
  _Float16* GiS  = (_Float16*)alloc(64 * 768 * 2);
  _Float16* GiA  = (_Float16*)alloc(64 * 768 * 2);
  _Float16* GiO  = (_Float16*)alloc(512 * 768 * 2);
  _Float16* GiH  = (_Float16*)alloc(64 * 768 * 2);
  _Float16* Hs16 = (_Float16*)alloc(64 * 256 * 2);
  float*    Hs32 = (float*)alloc(64 * 256 * 4);
  _Float16* Ha16 = (_Float16*)alloc(64 * 256 * 2);
  float*    Ha32 = (float*)alloc(64 * 256 * 4);
  _Float16* Ho16 = (_Float16*)alloc(512 * 256 * 2);
  float*    Ho32 = (float*)alloc(512 * 256 * 4);
  _Float16* hHd16 = (_Float16*)alloc(64 * 256 * 2);
  float*    hHd32 = (float*)alloc(64 * 256 * 4);
  _Float16* Xfc  = (_Float16*)alloc(64 * 512 * 2);
  _Float16* conc = (_Float16*)alloc(64 * 256 * 2);

  // --- weight conversions (f32 -> f16) ---
  struct { const float* s; _Float16* d; long n; } cv[] = {
    {f32w[0], wfih, 768L * 128}, {f32w[1], wfhh, 768L * 256},
    {f32w[4], wbih, 768L * 128}, {f32w[5], wbhh, 768L * 256},
    {f32w[8], wsih, 768L * 768}, {f32w[9], wshh, 768L * 256},
    {f32w[12], waih, 768L * 768}, {f32w[13], wahh, 768L * 256},
    {f32w[16], woih, 768L * 512}, {f32w[17], wohh, 768L * 256},
    {f32w[20], whih, 768L * 512}, {f32w[21], whhh, 768L * 256},
    {f32w[24], wfc1, 256L * 512},
  };
  for (auto& c : cv)
    cvt_f32_f16<<<dim3((unsigned)cdivl(c.n, 256)), 256, 0, stream>>>(c.s, c.d, c.n);

  // --- zero-init recurrent state (per-call; ws is not re-poisoned) ---
  hipMemsetAsync(hF16, 0, NBW * 256 * 2, stream);
  hipMemsetAsync(hB16, 0, NBW * 256 * 2, stream);
  hipMemsetAsync(hF32, 0, NBW * 256 * 4, stream);
  hipMemsetAsync(hB32, 0, NBW * 256 * 4, stream);
  hipMemsetAsync(A32,  0, (long)Bq * Rq * 256 * 4, stream);
  hipMemsetAsync(hHd16, 0, 64 * 256 * 2, stream);
  hipMemsetAsync(hHd32, 0, 64 * 256 * 4, stream);

  // --- embedding gather ---
  embed_gather<<<dim3((unsigned)cdivl(NTS * Eq, 256)), 256, 0, stream>>>(dig, emb, x16);

  // --- time-parallel input projections: Gi = x @ Wih^T + bih (both dirs) ---
  gemm_bias_f16<<<dim3((unsigned)(NTS / 16)), 256, 0, stream>>>(
      x16, wfih, f32w[2], GiF, (int)NTS, 768, 128, 0);
  gemm_bias_f16<<<dim3((unsigned)(NTS / 16)), 256, 0, stream>>>(
      x16, wbih, f32w[6], GiB, (int)NTS, 768, 128, 0);

  // --- bidirectional encoder scan (masked, length-aware) ---
  for (int t = 0; t < Sq; ++t) {
    int sF = t, sB = Sq - 1 - t;
    gru_step<<<dim3(40), 256, 0, stream>>>(GiF + (long)sF * 768, (long)Sq * 768,
                                           hF16, hF32, wfhh, f32w[3], lens, sF, (int)NBW);
    gru_step<<<dim3(40), 256, 0, stream>>>(GiB + (long)sB * 768, (long)Sq * 768,
                                           hB16, hB32, wbhh, f32w[7], lens, sB, (int)NBW);
  }

  // --- interaction scan over W turns + head GRU ---
  const long GATHER_TOT = 2L * 64 * 768 + 512L * 512 + 64L * 512 + 2L * 64 * 256 + 512L * 256;
  const long SCATTER_TOT = 2L * 64 * 256 + 512L * 256;
  for (int w = 0; w < Wq; ++w) {
    inter_gather<<<dim3((unsigned)cdivl(GATHER_TOT, 256)), 256, 0, stream>>>(
        hF32, hB32, A32, users, w, Xspk, Xadr, Xoth, Xhd,
        Hs16, Hs32, Ha16, Ha32, Ho16, Ho32);
    gemm_bias_f16<<<dim3(4), 256, 0, stream>>>(Xspk, wsih, f32w[10], GiS, 64, 768, 768, 0);
    gemm_bias_f16<<<dim3(4), 256, 0, stream>>>(Xadr, waih, f32w[14], GiA, 64, 768, 768, 0);
    gemm_bias_f16<<<dim3(32), 256, 0, stream>>>(Xoth, woih, f32w[18], GiO, 512, 768, 512, 0);
    gemm_bias_f16<<<dim3(4), 256, 0, stream>>>(Xhd, whih, f32w[22], GiH, 64, 768, 512, 0);
    gru_step<<<dim3(4), 256, 0, stream>>>(GiS, 768, Hs16, Hs32, wshh, f32w[11], nullptr, 0, 64);
    gru_step<<<dim3(4), 256, 0, stream>>>(GiA, 768, Ha16, Ha32, wahh, f32w[15], nullptr, 0, 64);
    gru_step<<<dim3(32), 256, 0, stream>>>(GiO, 768, Ho16, Ho32, wohh, f32w[19], nullptr, 0, 512);
    gru_step<<<dim3(4), 256, 0, stream>>>(GiH, 768, hHd16, hHd32, whhh, f32w[23], nullptr, 0, 64);
    inter_scatter<<<dim3((unsigned)cdivl(SCATTER_TOT, 256)), 256, 0, stream>>>(
        A32, users, w, Hs32, Ha32, Ho32);
  }

  // --- head: concated = tanh([A_res, ctx] @ fc1^T + b) ; scores ; fc2 ---
  fc_gather<<<dim3(128), 256, 0, stream>>>(A32, hHd32, resp, Xfc);
  gemm_bias_f16<<<dim3(4), 256, 0, stream>>>(Xfc, wfc1, f32w[25], conc, 64, 256, 512, 1);
  final_scores<<<dim3(64), 32, 0, stream>>>(A32, conc, resp, f32w[26], f32w[27], out);

  // --- emit A (second tuple element) after output (64x9) ---
  copy_f32<<<dim3((unsigned)cdivl((long)Bq * Rq * 256, 256)), 256, 0, stream>>>(
      A32, out + Bq * (Rq - 1), (long)Bq * Rq * 256);
}
